// ResBlock_24799141167428
// MI455X (gfx1250) — compile-verified, matrix operand order checked by reading, not run on hardware
//
#include <hip/hip_runtime.h>
#include <cstdint>
#include <cstddef>

typedef __attribute__((ext_vector_type(8)))  float  v8f;
typedef __attribute__((ext_vector_type(16))) __bf16 v16bf;
typedef __attribute__((vector_size(8))) int v2i;

#define ASG __attribute__((address_space(1)))
#define ASL __attribute__((address_space(3)))

#if defined(__AMDGCN__) && __has_builtin(__builtin_amdgcn_global_load_async_to_lds_b64)
#define HAVE_ASYNC 1
#else
#define HAVE_ASYNC 0
#endif

union FragBF { v16bf v; uint32_t u[8]; };

__device__ __forceinline__ int kpat(int j, int half) {
  return (j < 4 ? 2 * j : 16 + 2 * (j - 4)) + (half ? 8 : 0);
}
__device__ __forceinline__ float geluf(float x) {
  return 0.5f * x * (1.0f + erff(x * 0.7071067811865475f));
}
__device__ __forceinline__ uint32_t packbf2(float a, float b) {
  union { uint32_t u; __bf16 h[2]; } t;
  t.h[0] = (__bf16)a; t.h[1] = (__bf16)b; return t.u;
}
__device__ __forceinline__ v8f vzero() {
  v8f z;
#pragma unroll
  for (int i = 0; i < 8; ++i) z[i] = 0.f;
  return z;
}
__device__ __forceinline__ v8f wmma_bf16(const FragBF& a, const FragBF& b, v8f c) {
  return __builtin_amdgcn_wmma_f32_16x16x32_bf16(false, a.v, false, b.v, (short)0, c, false, false);
}
__device__ __forceinline__ int reflect(int v, int n) {
  if (v < 0) v = -v;
  if (v >= n) v = 2 * n - 2 - v;
  return v;
}
__device__ __forceinline__ void wait_async0() {
#if __has_builtin(__builtin_amdgcn_s_wait_asynccnt)
  __builtin_amdgcn_s_wait_asynccnt(0);
#else
  asm volatile("s_wait_asynccnt 0x0" ::: "memory");
#endif
}

// ---------------------------------------------------------------------------
// Weight pre-pack: f32 [KO][C][T] -> bf16 [tap][KO][C] (async-DMA friendly)
// ---------------------------------------------------------------------------
__global__ void k_pack(const float* __restrict__ w, __bf16* __restrict__ dst,
                       int KO, int C, int T) {
  int idx = blockIdx.x * 256 + threadIdx.x;
  int total = KO * C * T;
  if (idx >= total) return;
  int c = idx % C, r = idx / C;
  int k = r % KO, tap = r / KO;
  dst[idx] = (__bf16)w[((size_t)k * C + c) * T + tap];
}

// ---------------------------------------------------------------------------
// Implicit-GEMM conv (KSZ=3 pad 1, or KSZ=1 pad 0). Out channels = NT*16.
// 128 threads (4 waves), 64 pixels x (NT*16) channels per workgroup.
// Weights pre-packed bf16; B tiles staged with async global->LDS DMA.
// flags: bit0 = gelu on input, bit1 = gelu on output.
// ---------------------------------------------------------------------------
template <int NT, int KSZ>
__global__ void __launch_bounds__(128)
k_conv(const float* __restrict__ in, const __bf16* __restrict__ wpk,
       const float* __restrict__ bias, const float* __restrict__ res,
       float* __restrict__ out, int N, int C, int H, int W, int flags) {
  constexpr int KO = NT * 16;
  constexpr int AP = 36;        // bf16 pitch: 72B rows, 8B aligned, conflict-free
  constexpr int CP = KO + 4;
  extern __shared__ char smem[];
  __bf16* As = (__bf16*)smem;                                // [64][AP]
  __bf16* Bs = (__bf16*)(smem + 64 * AP * 2);                // [KO][AP]
  float*  Cs = (float*)(smem + 64 * AP * 2 + KO * AP * 2);   // [64][CP]

  const int tid = threadIdx.x;
  const int wave = tid >> 5, lane = tid & 31, half = lane >> 4, l16 = lane & 15;
  const int HW = H * W;
  const int n  = blockIdx.x / (HW >> 6);
  const int p0 = (blockIdx.x % (HW >> 6)) << 6;
  const int m   = tid & 63;   // pixel slot staged by this thread (loop invariant)
  const int ch0 = tid >> 6;   // channel parity
  const int p = p0 + m, py = p / W, px = p % W;
  const size_t gbase = (size_t)n * C * HW;

  v8f acc[NT];
#pragma unroll
  for (int t = 0; t < NT; ++t) acc[t] = vzero();

  const int PADP = (KSZ == 3) ? 1 : 0;
  for (int tap = 0; tap < KSZ * KSZ; ++tap) {
    const int dy = tap / KSZ - PADP, dx = tap % KSZ - PADP;
    const int yy = py + dy, xx = px + dx;
    const bool inb = (unsigned)yy < (unsigned)H && (unsigned)xx < (unsigned)W;
    const size_t goff = gbase + (size_t)yy * W + xx;
    for (int cc = 0; cc < C; cc += 32) {
      __syncthreads();
      // stage A: 64 pixels x 32 channels (im2col on the fly, zero pad)
#pragma unroll
      for (int i = 0; i < 16; ++i) {
        int c = 2 * i + ch0;
        float v = 0.f;
        if (inb) v = in[goff + (size_t)(cc + c) * HW];
        if (flags & 1) v = geluf(v);
        As[m * AP + c] = (__bf16)v;
      }
      if (inb && cc + 32 < C)
        __builtin_prefetch(in + goff + (size_t)(cc + 32 + ch0) * HW, 0, 0);
      // stage B (pre-packed bf16 [tap][k][c]) via async global->LDS DMA
#if HAVE_ASYNC
      for (int e = tid; e < KO * 8; e += 128) {
        int k = e >> 3, c0 = (e & 7) << 2;  // 4 bf16 = 8 bytes per issue
        __builtin_amdgcn_global_load_async_to_lds_b64(
            (ASG v2i*)(wpk + (size_t)(tap * KO + k) * C + cc + c0),
            (ASL v2i*)(Bs + k * AP + c0), 0, 0);
      }
      wait_async0();
#else
      for (int e = tid; e < KO * 16; e += 128) {
        int k = e >> 4, c0 = (e & 15) << 1;
        *(uint32_t*)(Bs + k * AP + c0) =
            *(const uint32_t*)(wpk + (size_t)(tap * KO + k) * C + cc + c0);
      }
#endif
      __syncthreads();
      FragBF fa;
      const __bf16* arow = As + (wave * 16 + l16) * AP;
#pragma unroll
      for (int j = 0; j < 8; ++j) fa.u[j] = *(const uint32_t*)(arow + kpat(j, half));
#pragma unroll
      for (int t = 0; t < NT; ++t) {
        FragBF fb;
        const __bf16* brow = Bs + (t * 16 + l16) * AP;
#pragma unroll
        for (int j = 0; j < 8; ++j) fb.u[j] = *(const uint32_t*)(brow + kpat(j, half));
        acc[t] = wmma_bf16(fa, fb, acc[t]);
      }
    }
  }
  __syncthreads();
#pragma unroll
  for (int t = 0; t < NT; ++t)
#pragma unroll
    for (int r = 0; r < 8; ++r)
      Cs[(wave * 16 + r + 8 * half) * CP + t * 16 + l16] = acc[t][r];
  __syncthreads();
#pragma unroll
  for (int j = 0; j < KO / 2; ++j) {
    int k = ch0 + 2 * j;
    float v = Cs[m * CP + k];
    if (bias) v += bias[k];
    if (flags & 2) v = geluf(v);
    size_t o = (((size_t)n * KO + k) * H + py) * W + px;
    if (res) v += res[o];
    out[o] = v;
  }
}

// ---------------------------------------------------------------------------
// Fused per-window spectral attention + memory-unit SE. One block per 8x8
// window, 128 threads (4 waves). LDS-resident QKV / attention / projection.
// Adds its result into asum (which already holds dyn_filter + x).
// ---------------------------------------------------------------------------
__global__ void __launch_bounds__(128)
k_se(const float* __restrict__ x, const float* __restrict__ qkvw,
     const float* __restrict__ qkvb, const float* __restrict__ apw,
     const float* __restrict__ apb, const float* __restrict__ ssw,
     const float* __restrict__ ssb, const float* __restrict__ mbp,
     const float* __restrict__ suw, const float* __restrict__ sub,
     float* __restrict__ asum, int N, int H, int W) {
  extern __shared__ char smem[];
  __bf16* QKV = (__bf16*)smem;            // [64][386] bf16 (49408B); reused as proj Bt
  __bf16* AX  = (__bf16*)(smem + 49408);  // [64][130] bf16 A matrix (16640B)
  __bf16* R2  = (__bf16*)(smem + 66048);  // qkv Bt chunks [96][130] -> SEout [64][130]
  __bf16* ATT = (__bf16*)(smem + 91008);  // 4 waves x [16][32] attn probs (4096B)
  float*  SCR = (float*)(smem + 95104);
  float* ys  = SCR;        // 128
  float* lrs = SCR + 128;  // 32
  float* fdv = SCR + 160;  // 256
  float* fdn = SCR + 416;  // 256
  float* zs  = SCR + 672;  // 32
  float* y2s = SCR + 704;  // 128

  const int tid = threadIdx.x, wave = tid >> 5, lane = tid & 31;
  const int half = lane >> 4, l16 = lane & 15;
  const int nw = W >> 3, nh = H >> 3;
  const int b = blockIdx.x;
  const int n = b / (nh * nw);
  const int rem = b % (nh * nw);
  const int wy = rem / nw, wx = rem % nw;
  const int y0p = wy * 8, x0p = wx * 8;

  // load window: AX[l][c] = x[n][c][wy*8+a][wx*8+bb], l = a*8+bb
  for (int idx = tid; idx < 64 * 128; idx += 128) {
    int c = idx & 127, l = idx >> 7;
    int a = l >> 3, bb = l & 7;
    AX[l * 130 + c] =
        (__bf16)x[(((size_t)n * 128 + c) * H + y0p + a) * W + x0p + bb];
  }
  // QKV = AX @ qkv_w + b, done in 4 groups of 96 output cols
  for (int ng = 0; ng < 4; ++ng) {
    __syncthreads();
    for (int idx = tid; idx < 96 * 128; idx += 128) {
      int k = idx & 127, nn = idx >> 7;
      R2[nn * 130 + k] = (__bf16)qkvw[(size_t)k * 384 + ng * 96 + nn];
    }
    __syncthreads();
    for (int t = wave; t < 24; t += 4) {
      int mt = t & 3, ntl = t >> 2;
      v8f accq = vzero();
      const __bf16* arow = AX + (mt * 16 + l16) * 130;
      const __bf16* brow = R2 + (ntl * 16 + l16) * 130;
      for (int kc = 0; kc < 4; ++kc) {
        FragBF fa, fb;
#pragma unroll
        for (int j = 0; j < 8; ++j) {
          int k0 = kc * 32 + kpat(j, half);
          fa.u[j] = *(const uint32_t*)(arow + k0);
          fb.u[j] = *(const uint32_t*)(brow + k0);
        }
        accq = wmma_bf16(fa, fb, accq);
      }
      int col = ng * 96 + ntl * 16 + l16;
      float bv = qkvb[col];
#pragma unroll
      for (int r = 0; r < 8; ++r) {
        int mm = mt * 16 + r + 8 * half;
        QKV[mm * 386 + col] = (__bf16)(accq[r] + bv);
      }
    }
  }
  __syncthreads();
  // zero pad region of attn-prob fragment (K = 16..31)
  {
    __bf16* aA = ATT + wave * 16 * 32;
#pragma unroll
    for (int r = 0; r < 8; ++r) aA[(r + 8 * half) * 32 + 16 + l16] = (__bf16)0.f;
  }
  __bf16* SEO = R2;  // attention output [64][130] bf16, reuses Bt region
  for (int hh = 0; hh < 2; ++hh) {
    const int hd = wave + 4 * hh;
    const int c0 = hd * 16;
    // L2 norms of q,k rows over L=64 (each lane owns d = lane%16)
    float sq = 0.f, sk = 0.f;
    for (int l = 0; l < 64; ++l) {
      float qv = (float)QKV[l * 386 + c0 + l16];
      float kv = (float)QKV[l * 386 + 128 + c0 + l16];
      sq += qv * qv; sk += kv * kv;
    }
    const float nqv = 1.f / (sqrtf(sq) + 1e-6f);
    const float nkv = 1.f / (sqrtf(sk) + 1e-6f);
    // attn = qn @ kn^T  (16x16, K = 64)
    v8f at = vzero();
    for (int kc = 0; kc < 2; ++kc) {
      FragBF fa, fb;
#pragma unroll
      for (int j = 0; j < 8; ++j) {
        int l0 = kc * 32 + kpat(j, half);
        fa.u[j] = packbf2((float)QKV[(l0)*386 + c0 + l16] * nqv,
                          (float)QKV[(l0 + 1) * 386 + c0 + l16] * nqv);
        fb.u[j] = packbf2((float)QKV[(l0)*386 + 128 + c0 + l16] * nkv,
                          (float)QKV[(l0 + 1) * 386 + 128 + c0 + l16] * nkv);
      }
      at = wmma_bf16(fa, fb, at);
    }
    // scaled softmax over e (16 lanes within each half-wave per row)
#pragma unroll
    for (int r = 0; r < 8; ++r) {
      float v = at[r] * 0.25f;
      float mx = v;
      for (int off = 8; off; off >>= 1) mx = fmaxf(mx, __shfl_xor(mx, off, 16));
      float e = __expf(v - mx);
      float s = e;
      for (int off = 8; off; off >>= 1) s += __shfl_xor(s, off, 16);
      at[r] = e / s;
    }
    __bf16* aA = ATT + wave * 16 * 32;
#pragma unroll
    for (int r = 0; r < 8; ++r) aA[(r + 8 * half) * 32 + l16] = (__bf16)at[r];
    __syncthreads();
    // out = attn @ v  (16 x 64, K = 16 zero-padded to 32)
    for (int nb = 0; nb < 4; ++nb) {
      FragBF fa, fb;
#pragma unroll
      for (int j = 0; j < 8; ++j) {
        int k0 = kpat(j, half);
        fa.u[j] = *(const uint32_t*)(aA + l16 * 32 + k0);
        if (j < 4) {
          int l = nb * 16 + l16;
          fb.u[j] = *(const uint32_t*)(QKV + l * 386 + 256 + c0 + k0);
        } else {
          fb.u[j] = 0u;
        }
      }
      v8f ov = wmma_bf16(fa, fb, vzero());
#pragma unroll
      for (int r = 0; r < 8; ++r) {
        int l = nb * 16 + l16;
        SEO[l * 130 + c0 + r + 8 * half] = (__bf16)ov[r];
      }
    }
    __syncthreads();
  }
  // memory-unit SE gating
  {
    float s = 0.f;
    for (int l = 0; l < 64; ++l) s += (float)SEO[l * 130 + tid];
    ys[tid] = s * (1.f / 64.f);
  }
  __syncthreads();
  if (tid < 32) {
    float s = ssb[tid];
    for (int c = 0; c < 128; ++c) s += ys[c] * ssw[c * 32 + tid];
    lrs[tid] = s;
  }
  __syncthreads();
  for (int mm = tid; mm < 256; mm += 128) {
    float s = 0.f;
    for (int j = 0; j < 32; ++j) s += lrs[j] * mbp[j * 256 + mm];
    fdv[mm] = s * 0.17677669529663687f;  // 32^-0.5
  }
  __syncthreads();
  {
    float mx = -1e30f;
    for (int mm = 0; mm < 256; ++mm) mx = fmaxf(mx, fdv[mm]);
    float ssum = 0.f;
    for (int mm = 0; mm < 256; ++mm) ssum += __expf(fdv[mm] - mx);
    for (int mm = tid; mm < 256; mm += 128) fdn[mm] = __expf(fdv[mm] - mx) / ssum;
  }
  __syncthreads();
  if (tid < 32) {
    float s = 0.f;
    for (int mm = 0; mm < 256; ++mm) s += fdn[mm] * mbp[tid * 256 + mm];
    zs[tid] = s;
  }
  __syncthreads();
  {
    float s = sub[tid];
    for (int j = 0; j < 32; ++j) s += zs[j] * suw[j * 128 + tid];
    y2s[tid] = 1.f / (1.f + __expf(-s));
  }
  __syncthreads();
  // gated A and projection GEMM: (out * y2) @ attn_proj_w + b, += asum
  for (int idx = tid; idx < 64 * 128; idx += 128) {
    int c = idx & 127, l = idx >> 7;
    AX[l * 130 + c] = (__bf16)((float)SEO[l * 130 + c] * y2s[c]);
  }
  __bf16* PB = QKV;  // reuse QKV region for proj Bt [128][130]
  for (int idx = tid; idx < 128 * 128; idx += 128) {
    int k = idx & 127, nn2 = idx >> 7;
    PB[nn2 * 130 + k] = (__bf16)apw[(size_t)k * 128 + nn2];
  }
  __syncthreads();
  for (int t = wave; t < 32; t += 4) {
    int mt = t & 3, ntl = t >> 2;
    v8f accp = vzero();
    const __bf16* arow = AX + (mt * 16 + l16) * 130;
    const __bf16* brow = PB + (ntl * 16 + l16) * 130;
    for (int kc = 0; kc < 4; ++kc) {
      FragBF fa, fb;
#pragma unroll
      for (int j = 0; j < 8; ++j) {
        int k0 = kc * 32 + kpat(j, half);
        fa.u[j] = *(const uint32_t*)(arow + k0);
        fb.u[j] = *(const uint32_t*)(brow + k0);
      }
      accp = wmma_bf16(fa, fb, accp);
    }
    int cOut = ntl * 16 + l16;
    float bv = apb[cOut];
#pragma unroll
    for (int r = 0; r < 8; ++r) {
      int l = mt * 16 + r + 8 * half;
      int a = l >> 3, bb = l & 7;
      size_t o = (((size_t)n * 128 + cOut) * H + y0p + a) * W + x0p + bb;
      asum[o] += accp[r] + bv;
    }
  }
}

// ---------------------------------------------------------------------------
// Elementwise helpers
// ---------------------------------------------------------------------------
__global__ void k_copy(const float* __restrict__ a, float* __restrict__ b, int n) {
  int i = blockIdx.x * 256 + threadIdx.x;
  if (i < n) b[i] = a[i];
}

__global__ void k_avgpool(const float* __restrict__ in, const float* __restrict__ addp,
                          float* __restrict__ out, int N, int Cc, int H, int W,
                          int k, int OH, int OW) {
  int idx = blockIdx.x * 256 + threadIdx.x;
  int total = N * Cc * OH * OW;
  if (idx >= total) return;
  int ox = idx % OW, t = idx / OW;
  int oy = t % OH; t /= OH;
  int c = t % Cc, n = t / Cc;
  const float* p = in + (((size_t)n * Cc + c) * H + oy * k) * W + ox * k;
  float s = 0.f;
  for (int i = 0; i < k; ++i)
    for (int j = 0; j < k; ++j) s += p[i * W + j];
  s *= 1.f / (float)(k * k);
  if (addp) s += addp[idx];
  out[idx] = s;
}

__global__ void k_resize(const float* __restrict__ src, float* __restrict__ dst,
                         int N, int Cc, int h, int w, int oh, int ow, int addFlag) {
  int idx = blockIdx.x * 256 + threadIdx.x;
  int total = N * Cc * oh * ow;
  if (idx >= total) return;
  int ox = idx % ow, t = idx / ow;
  int oy = t % oh; t /= oh;
  int c = t % Cc, n = t / Cc;
  float sy = oy * (float)(h - 1) / (float)(oh - 1);
  float sx = ox * (float)(w - 1) / (float)(ow - 1);
  int y0 = (int)floorf(sy), x0 = (int)floorf(sx);
  int y1 = y0 + 1; if (y1 > h - 1) y1 = h - 1;
  int x1 = x0 + 1; if (x1 > w - 1) x1 = w - 1;
  float wyf = sy - y0, wxf = sx - x0;
  const float* p = src + ((size_t)n * Cc + c) * h * w;
  float v = (p[y0 * w + x0] * (1.f - wxf) + p[y0 * w + x1] * wxf) * (1.f - wyf) +
            (p[y1 * w + x0] * (1.f - wxf) + p[y1 * w + x1] * wxf) * wyf;
  if (addFlag) dst[idx] += v; else dst[idx] = v;
}

__global__ void k_gap(const float* __restrict__ x, float* __restrict__ gap, int H, int W) {
  __shared__ float red[256];
  int nc = blockIdx.x;
  int HW = H * W;
  const float* p = x + (size_t)nc * HW;
  float s = 0.f;
  for (int i = threadIdx.x; i < HW; i += 256) s += p[i];
  red[threadIdx.x] = s;
  __syncthreads();
  for (int off = 128; off; off >>= 1) {
    if (threadIdx.x < off) red[threadIdx.x] += red[threadIdx.x + off];
    __syncthreads();
  }
  if (threadIdx.x == 0) gap[nc] = red[0] / (float)HW;
}

__global__ void k_lf(const float* __restrict__ gap, const float* __restrict__ dfw,
                     float* __restrict__ lf, int N) {
  int idx = blockIdx.x * 256 + threadIdx.x;
  if (idx >= N * 72) return;
  int o = idx % 72, n = idx / 72;
  float s = 0.f;
  for (int c = 0; c < 128; ++c) s += dfw[o * 128 + c] * gap[n * 128 + c];
  lf[idx] = tanhf(s);
}

__global__ void k_dyn(const float* __restrict__ x, const float* __restrict__ lf,
                      const float* __restrict__ gap, const float* __restrict__ ia,
                      const float* __restrict__ ll, const float* __restrict__ lh,
                      float* __restrict__ asum, int N, int H, int W, int d) {
  int idx = blockIdx.x * 256 + threadIdx.x;
  int total = N * 128 * H * W;
  if (idx >= total) return;
  int xc = idx % W, t = idx / W;
  int y = t % H; t /= H;
  int c = t % 128, n = t / 128;
  int g = c >> 4;
  const float* lfp = lf + n * 72 + g * 9;
  const float* base = x + ((size_t)n * 128 + c) * H * W;
  float low = 0.f;
#pragma unroll
  for (int i = 0; i < 3; ++i) {
    int ry = reflect(y + (i - 1) * d, H);
#pragma unroll
    for (int j = 0; j < 3; ++j) {
      int rx = reflect(xc + (j - 1) * d, W);
      low += base[ry * W + rx] * lfp[i * 3 + j];
    }
  }
  float xv = base[y * W + xc];
  float ga = gap[n * 128 + c];
  float iav = ia[c];
  asum[idx] = (low * (iav + 1.f) - iav * ga) * ll[c] + xv * (lh[c] + 1.f) + xv;
}

// ---------------------------------------------------------------------------
// Host orchestration
// ---------------------------------------------------------------------------
extern "C" void kernel_launch(void* const* d_in, const int* in_sizes, int n_in,
                              void* d_out, int out_size, void* d_ws, size_t ws_size,
                              hipStream_t stream) {
  (void)in_sizes; (void)n_in; (void)out_size; (void)ws_size;
  const float* x = (const float*)d_in[0];
  auto P = [&](int i) { return (const float*)d_in[i]; };

  float* ws = (float*)d_ws;
  float* h0    = ws;                  // 2*64*256*256 = 8388608
  float* resl  = ws + 8388608;
  float* yup   = ws + 16777216;       // <= 2*64*128*128
  float* xin   = ws + 18874368;
  float* tcv   = ws + 20971520;
  float* h1    = ws + 23068672;       // 2*128*128*128
  float* asum  = ws + 27262976;
  float* ybuf  = ws + 31457280;
  float* dpout = ws + 16777216;       // reuses loop scratch after the pyramid
  float* gap   = ws + 33554432;
  float* lfb   = ws + 33554432 + 256;
  __bf16* wpk  = (__bf16*)(ws + 33554944);  // packed bf16 weights (~467K halves)

  (void)hipFuncSetAttribute((const void*)k_se,
                            hipFuncAttributeMaxDynamicSharedMemorySize, 98432);

  const int N = 2, Hf = 256, Wf = 256;
  const size_t SM4 = 64 * 36 * 2 + 64 * 36 * 2 + 64 * 68 * 4;   // 26624
  const size_t SM8 = 64 * 36 * 2 + 128 * 36 * 2 + 64 * 132 * 4; // 47616

  // ---- pack all conv weights to bf16 [tap][kout][cin] ----
  const size_t OFF_BC1 = 0, OFF_BC2 = 36864, OFF_SUM = 73728;
  auto off_conv  = [](int i) { return (size_t)110592 + (size_t)i * 118784; };
  auto off_first = [&](int i) { return off_conv(i) + 36864; };
  auto off_proj  = [&](int i) { return off_conv(i) + 110592; };
  k_pack<<<dim3((36864 + 255) / 256), dim3(256), 0, stream>>>(P(1), wpk + OFF_BC1, 64, 64, 9);
  k_pack<<<dim3((36864 + 255) / 256), dim3(256), 0, stream>>>(P(3), wpk + OFF_BC2, 64, 64, 9);
  k_pack<<<dim3((36864 + 255) / 256), dim3(256), 0, stream>>>(P(5), wpk + OFF_SUM, 64, 64, 9);
  for (int i = 0; i < 3; ++i) {
    const int base = 7 + i * 17;
    k_pack<<<dim3((36864 + 255) / 256), dim3(256), 0, stream>>>(
        P(6 + i * 17), wpk + off_conv(i), 64, 64, 9);
    k_pack<<<dim3((73728 + 255) / 256), dim3(256), 0, stream>>>(
        P(base + 0), wpk + off_first(i), 128, 64, 9);
    k_pack<<<dim3((8192 + 255) / 256), dim3(256), 0, stream>>>(
        P(base + 1), wpk + off_proj(i), 64, 128, 1);
  }

  // h0 = gelu(conv3x3(x, bc1_w) + bc1_b)
  k_conv<4, 3><<<dim3(N * Hf * Wf / 64), dim3(128), SM4, stream>>>(
      x, wpk + OFF_BC1, P(2), nullptr, h0, N, 64, Hf, Wf, 2);
  k_copy<<<dim3(8388608 / 256), dim3(256), 0, stream>>>(h0, resl, 8388608);

  const int pools[3] = {8, 4, 2};
  const int dils[3] = {3, 7, 9};
  for (int i = 0; i < 3; ++i) {
    const int Hs = Hf / pools[i], Ws_ = Wf / pools[i];
    const int base = 7 + i * 17;

    int tot = N * 64 * Hs * Ws_;
    k_avgpool<<<dim3((tot + 255) / 256), dim3(256), 0, stream>>>(
        h0, (i > 0) ? yup : nullptr, xin, N, 64, Hf, Wf, pools[i], Hs, Ws_);
    k_conv<4, 3><<<dim3(N * Hs * Ws_ / 64), dim3(128), SM4, stream>>>(
        xin, wpk + off_conv(i), nullptr, nullptr, tcv, N, 64, Hs, Ws_, 0);
    // first_w: 64 -> 128
    k_conv<8, 3><<<dim3(N * Hs * Ws_ / 64), dim3(128), SM8, stream>>>(
        tcv, wpk + off_first(i), nullptr, nullptr, h1, N, 64, Hs, Ws_, 0);
    k_gap<<<dim3(N * 128), dim3(256), 0, stream>>>(h1, gap, Hs, Ws_);
    k_lf<<<dim3(1), dim3(256), 0, stream>>>(gap, P(base + 3), lfb, N);
    int tot2 = N * 128 * Hs * Ws_;
    k_dyn<<<dim3((tot2 + 255) / 256), dim3(256), 0, stream>>>(
        h1, lfb, gap, P(base + 6), P(base + 4), P(base + 5), asum, N, Hs, Ws_,
        dils[i]);
    k_se<<<dim3(N * (Hs / 8) * (Ws_ / 8)), dim3(128), 98432, stream>>>(
        h1, P(base + 7), P(base + 8), P(base + 9), P(base + 10), P(base + 11),
        P(base + 12), P(base + 13), P(base + 14), P(base + 15), asum, N, Hs, Ws_);
    // proj 1x1: 128 -> 64, + residual (dual_branch input)
    k_conv<4, 1><<<dim3(N * Hs * Ws_ / 64), dim3(128), SM4, stream>>>(
        asum, wpk + off_proj(i), P(base + 2), tcv, ybuf, N, 128, Hs, Ws_, 0);
    int tot3 = N * 64 * Hf * Wf;
    k_resize<<<dim3((tot3 + 255) / 256), dim3(256), 0, stream>>>(
        ybuf, resl, N, 64, Hs, Ws_, Hf, Wf, 1);
    if (i < 2) {
      int tot4 = N * 64 * (2 * Hs) * (2 * Ws_);
      k_resize<<<dim3((tot4 + 255) / 256), dim3(256), 0, stream>>>(
          ybuf, yup, N, 64, Hs, Ws_, 2 * Hs, 2 * Ws_, 0);
    }
  }
  // dpout = conv3x3(gelu(resl), sum_w)
  k_conv<4, 3><<<dim3(N * Hf * Wf / 64), dim3(128), SM4, stream>>>(
      resl, wpk + OFF_SUM, nullptr, nullptr, dpout, N, 64, Hf, Wf, 1);
  // d_out = conv3x3(dpout, bc2_w) + bc2_b + x
  k_conv<4, 3><<<dim3(N * Hf * Wf / 64), dim3(128), SM4, stream>>>(
      dpout, wpk + OFF_BC2, P(4), x, (float*)d_out, N, 64, Hf, Wf, 0);
}
